// SwinBasedFusion_67138928771163
// MI455X (gfx1250) — compile-verified, matrix operand order checked by reading, not run on hardware
//
#include <hip/hip_runtime.h>
#include <math.h>

// ---------------- types ----------------
typedef __bf16 bf16;
typedef __attribute__((ext_vector_type(8)))  __bf16 bf16x8;
typedef __attribute__((ext_vector_type(16))) __bf16 bf16x16;
typedef __attribute__((ext_vector_type(8)))  float  f32x8;

#define T_TOK 131072   // B*H*W = 2*256*256

enum { EP_STORE_BF16 = 0, EP_LRELU_BF16 = 1, EP_GELU_BF16 = 2, EP_STORE_F32 = 3, EP_ACCUM_F32 = 4 };

// Load one 16-bit A/B fragment (16 bf16 per lane) from an LDS row.
// CDNA5 16-bit A layout: lanes 0-15 hold K {0..7, 16..23}, lanes 16-31 hold K {8..15, 24..31}.
__device__ __forceinline__ bf16x16 frag_ld(const bf16* rowp, int half) {
    const bf16x8* q = reinterpret_cast<const bf16x8*>(rowp);
    bf16x8 lo = q[half];       // K = half*8 .. half*8+7
    bf16x8 hi = q[2 + half];   // K = 16 + half*8 ..
    bf16x16 r;
#pragma unroll
    for (int i = 0; i < 8; ++i) { r[i] = lo[i]; r[8 + i] = hi[i]; }
    return r;
}

// CDNA5 async Memory->LDS copy (ASYNCcnt-tracked, no VGPR round trip).
// LDS destination address = low 32 bits of the generic pointer (ISA 10.2 aperture rule).
__device__ __forceinline__ void async_ld128(void* lds_ptr, const void* gaddr) {
    unsigned lds_off = (unsigned)(uintptr_t)lds_ptr;
    asm volatile("global_load_async_to_lds_b128 %0, %1, off"
                 :: "v"(lds_off), "v"((unsigned long long)(uintptr_t)gaddr)
                 : "memory");
}
__device__ __forceinline__ void wait_async0() {
    asm volatile("s_wait_asynccnt 0x0" ::: "memory");
}

// ---------------- generic WMMA GEMM: Out = epilogue(A[MxK] @ Wt[NxK]^T + bias) ----------------
// Block tile 128x128, 8 waves (2 M x 4 N), each wave 64x32 = 8 wmma accumulators.
__global__ __launch_bounds__(256)
void wmma_gemm(const bf16* __restrict__ A, const bf16* __restrict__ Wt,
               const float* __restrict__ bias, void* __restrict__ Out,
               int M, int N, int K, int mode)
{
    __shared__ bf16 As[128][40];   // 32 cols + 8 pad (rows stay 16B aligned)
    __shared__ bf16 Bs[128][40];

    const int tid  = threadIdx.x;
    const int bm   = blockIdx.y * 128;
    const int bn   = blockIdx.x * 128;
    const int wave = tid >> 5, lane = tid & 31;
    const int wm   = wave >> 2, wn  = wave & 3;    // 2x4 wave grid
    const int half = lane >> 4, l   = lane & 15;

    f32x8 acc[4][2] = {};

    for (int k0 = 0; k0 < K; k0 += 32) {
        __syncthreads();   // previous iteration's fragment reads complete
#pragma unroll
        for (int i = 0; i < 2; ++i) {               // A tile: 128x32 bf16, 512 x 16B chunks
            int idx = tid + 256 * i;
            int row = idx >> 2, c8 = (idx & 3) * 8;
            async_ld128(&As[row][c8], A + (size_t)(bm + row) * K + k0 + c8);
        }
#pragma unroll
        for (int i = 0; i < 2; ++i) {               // B tile: 128x32 bf16
            int idx = tid + 256 * i;
            int row = idx >> 2, c8 = (idx & 3) * 8;
            async_ld128(&Bs[row][c8], Wt + (size_t)(bn + row) * K + k0 + c8);
        }
        wait_async0();
        __syncthreads();   // all waves' tiles landed in LDS

        bf16x16 bfrag[2];
#pragma unroll
        for (int nt = 0; nt < 2; ++nt) bfrag[nt] = frag_ld(&Bs[wn * 32 + nt * 16 + l][0], half);
#pragma unroll
        for (int mt = 0; mt < 4; ++mt) {
            bf16x16 afrag = frag_ld(&As[wm * 64 + mt * 16 + l][0], half);
#pragma unroll
            for (int nt = 0; nt < 2; ++nt)
                acc[mt][nt] = __builtin_amdgcn_wmma_f32_16x16x32_bf16(
                    false, afrag, false, bfrag[nt], (short)0, acc[mt][nt], false, false);
        }
    }

#pragma unroll
    for (int mt = 0; mt < 4; ++mt)
#pragma unroll
        for (int nt = 0; nt < 2; ++nt) {
            int col = bn + wn * 32 + nt * 16 + l;
            float bv = bias ? bias[col] : 0.0f;
#pragma unroll
            for (int r = 0; r < 8; ++r) {
                int row = bm + wm * 64 + mt * 16 + half * 8 + r;
                float v = acc[mt][nt][r] + bv;
                size_t off = (size_t)row * N + col;
                if (mode == EP_LRELU_BF16) {
                    v = v > 0.0f ? v : 0.01f * v;
                    ((bf16*)Out)[off] = (bf16)v;
                } else if (mode == EP_GELU_BF16) {
                    v = 0.5f * v * (1.0f + erff(v * 0.70710678118654752f));
                    ((bf16*)Out)[off] = (bf16)v;
                } else if (mode == EP_STORE_BF16) {
                    ((bf16*)Out)[off] = (bf16)v;
                } else if (mode == EP_STORE_F32) {
                    ((float*)Out)[off] = v;
                } else {  // EP_ACCUM_F32 (residual add)
                    ((float*)Out)[off] += v;
                }
            }
        }
}

// ---------------- LayerNorm (per token row), writes bf16; supports in-place ----------------
template <typename Tin>
__global__ __launch_bounds__(256)
void ln_kernel(const Tin* __restrict__ X, const float* __restrict__ g,
               const float* __restrict__ b, bf16* __restrict__ Y, int C)
{
    __shared__ float red[256];
    const size_t t = blockIdx.x;
    const Tin* xr = X + t * (size_t)C;
    bf16* yr = Y + t * (size_t)C;
    const int tid = threadIdx.x;
    const int vpt = C >> 8;  // 1 (C=256) or 8 (C=2048)

    float v[8];
    float s = 0.0f;
    for (int i = 0; i < vpt; ++i) { v[i] = (float)xr[tid + (i << 8)]; s += v[i]; }
    red[tid] = s; __syncthreads();
    for (int o = 128; o; o >>= 1) { if (tid < o) red[tid] += red[tid + o]; __syncthreads(); }
    float mean = red[0] / (float)C;
    __syncthreads();
    float ss = 0.0f;
    for (int i = 0; i < vpt; ++i) { float d = v[i] - mean; ss += d * d; }
    red[tid] = ss; __syncthreads();
    for (int o = 128; o; o >>= 1) { if (tid < o) red[tid] += red[tid + o]; __syncthreads(); }
    float rs = rsqrtf(red[0] / (float)C + 1e-5f);
    for (int i = 0; i < vpt; ++i) {
        int c = tid + (i << 8);
        yr[c] = (bf16)((v[i] - mean) * rs * g[c] + b[c]);
    }
}

// ---------------- windowed attention: one block per (window, head) ----------------
__device__ __forceinline__ int swin_region(int h) {
    // slices for shift=4 on a 256 canvas: [0,248) / [248,252) / [252,256)
    return (h < 248) ? 0 : (h < 252 ? 1 : 2);
}

__global__ __launch_bounds__(128)
void win_attn_kernel(const bf16* __restrict__ qkv, const float* __restrict__ rpb,
                     bf16* __restrict__ attnout, int shift)
{
    const int id   = blockIdx.x;
    const int head = id & 7;
    const int w    = id >> 3;
    const int bw   = w >> 10;          // batch
    const int wi   = w & 1023;
    const int wy   = wi >> 5, wx = wi & 31;

    __shared__ bf16 Qs[64][40];
    __shared__ bf16 Ks[64][40];
    __shared__ bf16 Vt[32][72];        // transposed V: [dim][kv]
    __shared__ bf16 Ps[64][72];        // softmax probs
    __shared__ int  toks[64];

    const int tid = threadIdx.x;
    if (tid < 64) {
        int py = tid >> 3, px = tid & 7;
        int hh = (wy * 8 + py + shift) & 255;
        int ww = (wx * 8 + px + shift) & 255;
        toks[tid] = ((bw << 8) + hh) * 256 + ww;
    }
    __syncthreads();

    // stage Q, K (row-major) and V (transposed) into LDS: 64 pos x 32 dims each
#pragma unroll
    for (int i = 0; i < 2; ++i) {
        int idx = tid + 128 * i;
        int p = idx >> 2, c8 = (idx & 3) * 8;
        size_t base = (size_t)toks[p] * 768 + head * 32 + c8;
        bf16x8 qv = *reinterpret_cast<const bf16x8*>(qkv + base);
        bf16x8 kv = *reinterpret_cast<const bf16x8*>(qkv + base + 256);
        bf16x8 vv = *reinterpret_cast<const bf16x8*>(qkv + base + 512);
        *reinterpret_cast<bf16x8*>(&Qs[p][c8]) = qv;
        *reinterpret_cast<bf16x8*>(&Ks[p][c8]) = kv;
#pragma unroll
        for (int j = 0; j < 8; ++j) Vt[c8 + j][p] = vv[j];
    }
    __syncthreads();

    const int wv = tid >> 5, lane = tid & 31, half = lane >> 4, l = lane & 15;

    // S = Q Kt : each wave does rows [wv*16, wv*16+16), K = 32 -> one WMMA per 16x16 tile
    f32x8 sacc[4] = {};
    bf16x16 afrag = frag_ld(&Qs[wv * 16 + l][0], half);
#pragma unroll
    for (int nt = 0; nt < 4; ++nt) {
        bf16x16 bfrag = frag_ld(&Ks[nt * 16 + l][0], half);
        sacc[nt] = __builtin_amdgcn_wmma_f32_16x16x32_bf16(
            false, afrag, false, bfrag, (short)0, sacc[nt], false, false);
    }

    // scale + rel-pos bias + (optional) shift mask, then softmax per query row
    const float SCALE = 0.17677669529663687f;  // 32^-0.5
    float vals[4][8];
#pragma unroll
    for (int nt = 0; nt < 4; ++nt) {
        int kpos = nt * 16 + l;
        int ky = kpos >> 3, kx = kpos & 7;
#pragma unroll
        for (int r = 0; r < 8; ++r) {
            int qpos = wv * 16 + half * 8 + r;
            int qy = qpos >> 3, qx = qpos & 7;
            int rel = (qy - ky + 7) * 15 + (qx - kx + 7);
            float s = sacc[nt][r] * SCALE + rpb[rel * 8 + head];
            if (shift > 0) {
                int rq = swin_region(wy * 8 + qy) * 3 + swin_region(wx * 8 + qx);
                int rk = swin_region(wy * 8 + ky) * 3 + swin_region(wx * 8 + kx);
                if (rq != rk) s -= 100.0f;
            }
            vals[nt][r] = s;
        }
    }

    // row softmax: each row lives across 16 lanes (fixed half) x 4 accums
#pragma unroll
    for (int r = 0; r < 8; ++r) {
        float m = fmaxf(fmaxf(vals[0][r], vals[1][r]), fmaxf(vals[2][r], vals[3][r]));
        for (int d = 1; d < 16; d <<= 1) m = fmaxf(m, __shfl_xor(m, d, 32));
        float ssum = 0.0f;
#pragma unroll
        for (int nt = 0; nt < 4; ++nt) { float e = __expf(vals[nt][r] - m); vals[nt][r] = e; ssum += e; }
        for (int d = 1; d < 16; d <<= 1) ssum += __shfl_xor(ssum, d, 32);
        float inv = 1.0f / ssum;
        int qrow = wv * 16 + half * 8 + r;
#pragma unroll
        for (int nt = 0; nt < 4; ++nt) Ps[qrow][nt * 16 + l] = (bf16)(vals[nt][r] * inv);
    }
    __syncthreads();

    // O = P V : K = 64 -> 2 WMMA per 16x16 output tile, 2 dim-tiles
    f32x8 oacc[2] = {};
#pragma unroll
    for (int kt = 0; kt < 2; ++kt) {
        bf16x16 pa = frag_ld(&Ps[wv * 16 + l][kt * 32], half);
#pragma unroll
        for (int nt = 0; nt < 2; ++nt) {
            bf16x16 vb = frag_ld(&Vt[nt * 16 + l][kt * 32], half);
            oacc[nt] = __builtin_amdgcn_wmma_f32_16x16x32_bf16(
                false, pa, false, vb, (short)0, oacc[nt], false, false);
        }
    }
#pragma unroll
    for (int nt = 0; nt < 2; ++nt)
#pragma unroll
        for (int r = 0; r < 8; ++r) {
            int qpos = wv * 16 + half * 8 + r;
            int d = nt * 16 + l;
            attnout[(size_t)toks[qpos] * 256 + head * 32 + d] = (bf16)oacc[nt][r];
        }
}

// ---------------- elementwise helpers ----------------
__global__ void f32_to_bf16_kernel(const float* __restrict__ in, bf16* __restrict__ out, int n) {
    int i = blockIdx.x * 256 + threadIdx.x;
    if (i < n) out[i] = (bf16)in[i];
}

__global__ void concat_bf16_kernel(const float* __restrict__ pcd, const float* __restrict__ img,
                                   bf16* __restrict__ xin) {
    size_t i = (size_t)blockIdx.x * 256 + threadIdx.x;   // over T*512
    size_t t = i >> 9;
    int c = (int)(i & 511);
    float v = (c < 256) ? pcd[t * 256 + c] : img[t * 256 + (c - 256)];
    xin[i] = (bf16)v;
}

__global__ void final_add_kernel(const float* __restrict__ x, const float* __restrict__ pcd,
                                 float* __restrict__ out) {
    size_t i = (size_t)blockIdx.x * 256 + threadIdx.x;   // over T*256
    out[i] = x[i] + pcd[i];
}

// ---------------- host orchestration ----------------
extern "C" void kernel_launch(void* const* d_in, const int* in_sizes, int n_in,
                              void* d_out, int out_size, void* d_ws, size_t ws_size,
                              hipStream_t stream) {
    (void)in_sizes; (void)n_in; (void)out_size; (void)ws_size;

    const float* pcd    = (const float*)d_in[0];
    const float* img    = (const float*)d_in[1];
    const float* rd_w1  = (const float*)d_in[2];
    const float* rd_b1  = (const float*)d_in[3];
    const float* rd_lng = (const float*)d_in[4];
    const float* rd_lnb = (const float*)d_in[5];
    const float* rd_w2  = (const float*)d_in[6];
    const float* rd_b2  = (const float*)d_in[7];
    const float* n1g    = (const float*)d_in[8];
    const float* n1b    = (const float*)d_in[9];
    const float* qkv_w  = (const float*)d_in[10];
    const float* qkv_b  = (const float*)d_in[11];
    const float* rpb    = (const float*)d_in[12];
    const float* proj_w = (const float*)d_in[13];
    const float* proj_b = (const float*)d_in[14];
    const float* n2g    = (const float*)d_in[15];
    const float* n2b    = (const float*)d_in[16];
    const float* mlp_w1 = (const float*)d_in[17];
    const float* mlp_b1 = (const float*)d_in[18];
    const float* mlp_w2 = (const float*)d_in[19];
    const float* mlp_b2 = (const float*)d_in[20];

    char* ws = (char*)d_ws;
    const size_t T = T_TOK;
    // bf16 weight copies
    size_t o_w1 = 0;
    size_t o_w2 = o_w1 + (size_t)2048 * 512 * 2;
    size_t o_wq = o_w2 + (size_t)256 * 2048 * 2;
    size_t o_wp = o_wq + (size_t)2 * 768 * 256 * 2;
    size_t o_m1 = o_wp + (size_t)2 * 256 * 256 * 2;
    size_t o_m2 = o_m1 + (size_t)2 * 1024 * 256 * 2;
    size_t o_x   = o_m2 + (size_t)2 * 256 * 1024 * 2;   // f32 residual stream  [T x 256]
    size_t o_xbf = o_x   + T * 256 * 4;                 // bf16 LN(x)           [T x 256]
    size_t o_xin = o_xbf + T * 256 * 2;                 // bf16 concat input    [T x 512]
    size_t o_h   = o_xin + T * 512 * 2;                 // bf16 hidden          [T x 2048], later reused
    size_t o_qkv = o_h;                                 // bf16 [T x 768]
    size_t o_att = o_h + T * 768 * 2;                   // bf16 [T x 256]
    size_t o_hid = o_att + T * 256 * 2;                 // bf16 [T x 1024]

    bf16* w1bf  = (bf16*)(ws + o_w1);
    bf16* w2bf  = (bf16*)(ws + o_w2);
    bf16* wqbf  = (bf16*)(ws + o_wq);
    bf16* wpbf  = (bf16*)(ws + o_wp);
    bf16* m1bf  = (bf16*)(ws + o_m1);
    bf16* m2bf  = (bf16*)(ws + o_m2);
    float* x    = (float*)(ws + o_x);
    bf16* xbf   = (bf16*)(ws + o_xbf);
    bf16* xin   = (bf16*)(ws + o_xin);
    bf16* h     = (bf16*)(ws + o_h);
    bf16* qkvb  = (bf16*)(ws + o_qkv);
    bf16* attn  = (bf16*)(ws + o_att);
    bf16* hid   = (bf16*)(ws + o_hid);

    auto cvt = [&](const float* src, bf16* dst, int n) {
        f32_to_bf16_kernel<<<(n + 255) / 256, 256, 0, stream>>>(src, dst, n);
    };
    cvt(rd_w1, w1bf, 2048 * 512);
    cvt(rd_w2, w2bf, 256 * 2048);
    cvt(qkv_w, wqbf, 2 * 768 * 256);
    cvt(proj_w, wpbf, 2 * 256 * 256);
    cvt(mlp_w1, m1bf, 2 * 1024 * 256);
    cvt(mlp_w2, m2bf, 2 * 256 * 1024);

    // reduce MLP: concat -> gemm1(leakyReLU) -> LN(in-place) -> gemm2 -> x (f32)
    concat_bf16_kernel<<<(unsigned)(T * 512 / 256), 256, 0, stream>>>(pcd, img, xin);
    wmma_gemm<<<dim3(2048 / 128, (unsigned)(T / 128)), 256, 0, stream>>>(
        xin, w1bf, rd_b1, h, (int)T, 2048, 512, EP_LRELU_BF16);
    ln_kernel<bf16><<<(unsigned)T, 256, 0, stream>>>(h, rd_lng, rd_lnb, h, 2048);
    wmma_gemm<<<dim3(256 / 128, (unsigned)(T / 128)), 256, 0, stream>>>(
        h, w2bf, rd_b2, x, (int)T, 256, 2048, EP_STORE_F32);

    for (int layer = 0; layer < 2; ++layer) {
        int shift = (layer & 1) ? 4 : 0;
        // attention branch
        ln_kernel<float><<<(unsigned)T, 256, 0, stream>>>(x, n1g + layer * 256, n1b + layer * 256, xbf, 256);
        wmma_gemm<<<dim3(768 / 128, (unsigned)(T / 128)), 256, 0, stream>>>(
            xbf, wqbf + (size_t)layer * 768 * 256, qkv_b + layer * 768, qkvb, (int)T, 768, 256, EP_STORE_BF16);
        win_attn_kernel<<<2048 * 8, 128, 0, stream>>>(qkvb, rpb + (size_t)layer * 225 * 8, attn, shift);
        wmma_gemm<<<dim3(256 / 128, (unsigned)(T / 128)), 256, 0, stream>>>(
            attn, wpbf + (size_t)layer * 256 * 256, proj_b + layer * 256, x, (int)T, 256, 256, EP_ACCUM_F32);
        // MLP branch
        ln_kernel<float><<<(unsigned)T, 256, 0, stream>>>(x, n2g + layer * 256, n2b + layer * 256, xbf, 256);
        wmma_gemm<<<dim3(1024 / 128, (unsigned)(T / 128)), 256, 0, stream>>>(
            xbf, m1bf + (size_t)layer * 1024 * 256, mlp_b1 + layer * 1024, hid, (int)T, 1024, 256, EP_GELU_BF16);
        wmma_gemm<<<dim3(256 / 128, (unsigned)(T / 128)), 256, 0, stream>>>(
            hid, m2bf + (size_t)layer * 256 * 1024, mlp_b2 + layer * 256, x, (int)T, 256, 1024, EP_ACCUM_F32);
    }

    final_add_kernel<<<(unsigned)(T * 256 / 256), 256, 0, stream>>>(x, pcd, (float*)d_out);
}
